// GATEncoder_21921513079382
// MI455X (gfx1250) — matrix operand order repaired and checked by reading, not compiled, not run on hardware
//
#include <hip/hip_runtime.h>
#include <math.h>

// ---------------------------------------------------------------------------
// GATv2 encoder for MI455X (gfx1250, wave32).
//   layer1: GATv2(in=2 -> 4 heads x 32) -> mean heads -> LN -> ELU
//   layer2: GATv2(32 -> 1 head x 16)    -> LN
// Dense node transforms: v_wmma_f32_16x16x32_f16, B tiles + bias staged in LDS,
// branchless epilogue (clamped loads, padded output regions).
// Edge phases: float4 gathers + global f32 atomics (L2-resident working set).
// ---------------------------------------------------------------------------

typedef __attribute__((ext_vector_type(16))) _Float16 v16h;
typedef __attribute__((ext_vector_type(8)))  float    v8f;

#define NEG_SLOPE 0.2f
#define LN_EPS    1e-5f
#define ROWPAD    16          // output-region row padding for branchless stores

#if defined(__has_builtin)
#if __has_builtin(__hip_atomic_fetch_max)
#define HAVE_HIP_FMAX 1
#endif
#endif

__device__ __forceinline__ void atomic_max_float(float* addr, float val) {
#ifdef HAVE_HIP_FMAX
  __hip_atomic_fetch_max(addr, val, __ATOMIC_RELAXED, __HIP_MEMORY_SCOPE_AGENT);
#else
  // classic signed/unsigned trick; requires init to -inf
  if (val >= 0.0f) atomicMax((int*)addr, __float_as_int(val));
  else             atomicMin((unsigned int*)addr, __float_as_uint(val));
#endif
}

__global__ void fill_f32(float* __restrict__ p, float v, long n) {
  long i = (long)blockIdx.x * blockDim.x + threadIdx.x;
  if (i < n) p[i] = v;
}

// ---------------------------------------------------------------------------
// Y[n x FOUT] = X[n x FIN] @ W[FIN x FOUT] + b
// One wave per 16-row tile of X. FIN in {2,32}, FOUT % 16 == 0.
// A layout (16-bit A 16x32): lane l -> row = l%16, half h = l/16;
//   elem j<8  -> k = 8h + j        (contiguous run -> float4 loads)
//   elem j>=8 -> k = 16 + 8h + j-8 (contiguous run -> float4 loads)
// B layout (16-bit B 32x16): lane l -> col = l%16; elem j -> same k as A elem j (8h+j / 16+8h+j-8)
//   -> pre-packed per (tile, lane) into LDS: each lane reads 32B contiguous.
// D layout (f32 C/D 16x16): reg r -> row = r + 8h, col = l%16
// Branchless: load rows clamped to n-1; Y has >= ROWPAD padding rows, so the
// tail tile stores unconditionally into padding.
// ---------------------------------------------------------------------------
template <int FIN, int FOUT>
__global__ void xform_wmma(const float* __restrict__ X, const float* __restrict__ W,
                           const float* __restrict__ bvec, float* __restrict__ Y, int n) {
  static_assert((FIN == 2 || FIN == 32) && (FOUT % 16) == 0, "bad dims");
  constexpr int NT = FOUT / 16;                       // # of 16-col tiles
  __shared__ __attribute__((aligned(32))) _Float16 bl[NT * 32 * 16];
  __shared__ float bsh[FOUT];

  // cooperative pre-pack of B (f32 -> f16, WMMA lane order) + bias into LDS
  for (int idx = threadIdx.x; idx < NT * 32 * 16; idx += blockDim.x) {
    int j    = idx & 15;
    int ln   = (idx >> 4) & 31;
    int tile = idx >> 9;
    int k    = (j < 8) ? (8 * (ln >> 4) + j) : (16 + 8 * (ln >> 4) + (j - 8));
    int col  = tile * 16 + (ln & 15);
    float wv = (k < FIN) ? W[(size_t)k * FOUT + col] : 0.0f;
    bl[idx] = (_Float16)wv;
  }
  for (int idx = threadIdx.x; idx < FOUT; idx += blockDim.x) bsh[idx] = bvec[idx];
  __syncthreads();

  int wave = (int)((blockIdx.x * (long)blockDim.x + threadIdx.x) >> 5);
  int lane = threadIdx.x & 31;
  int m0 = wave * 16;
  if (m0 >= n) return;                 // wave-uniform (after barrier): EXEC all-1 for WMMA
  int m  = lane & 15;
  int hh = lane >> 4;
  int row = m0 + m;
  row = (row < n) ? row : (n - 1);     // clamp: branchless in-bounds loads

  // ---- pack A (contiguous K-runs -> vector loads) ----
  v16h a;
  if constexpr (FIN == 2) {
#pragma unroll
    for (int j = 0; j < 16; ++j) a[j] = (_Float16)0.0f;
    float2 xv = *(const float2*)(X + (size_t)row * 2);
    if (hh == 0) { a[0] = (_Float16)xv.x; a[1] = (_Float16)xv.y; }
  } else {  // FIN == 32
    const float4* px = (const float4*)(X + (size_t)row * 32);
    float4 q0 = px[2 * hh];     float4 q1 = px[2 * hh + 1];      // k = 8h .. 8h+7
    float4 q2 = px[4 + 2 * hh]; float4 q3 = px[4 + 2 * hh + 1];  // k = 16+8h .. +7
    a[0] = (_Float16)q0.x;  a[1] = (_Float16)q0.y;  a[2]  = (_Float16)q0.z;  a[3]  = (_Float16)q0.w;
    a[4] = (_Float16)q1.x;  a[5] = (_Float16)q1.y;  a[6]  = (_Float16)q1.z;  a[7]  = (_Float16)q1.w;
    a[8] = (_Float16)q2.x;  a[9] = (_Float16)q2.y;  a[10] = (_Float16)q2.z;  a[11] = (_Float16)q2.w;
    a[12] = (_Float16)q3.x; a[13] = (_Float16)q3.y; a[14]  = (_Float16)q3.z; a[15] = (_Float16)q3.w;
  }

  const v16h* bp = (const v16h*)bl;
  float* Yt = Y + (size_t)m0 * FOUT;   // tile base; padding rows absorb the tail
#pragma unroll
  for (int t = 0; t < NT; ++t) {
    int col = t * 16 + (lane & 15);
    v16h b = bp[t * 32 + lane];        // 32B contiguous from LDS (2x ds_load_b128)
    v8f c = {};
    c = __builtin_amdgcn_wmma_f32_16x16x32_f16(
        /*neg_a=*/false, a, /*neg_b=*/false, b,
        /*c_mod=*/(short)0, c, /*reuse_a=*/false, /*reuse_b=*/false);
    float bias = bsh[col];
#pragma unroll
    for (int r = 0; r < 8; ++r)
      Yt[(size_t)(r + 8 * hh) * FOUT + col] = c[r] + bias;   // unconditional
  }
}

// ------------------------- layer 1 edge phases (H=4, C=32) -----------------
__global__ void edge_logit1(const float* __restrict__ xl, const float* __restrict__ xr,
                            const float* __restrict__ ea, const int* __restrict__ src,
                            const int* __restrict__ dst, const float* __restrict__ We,
                            const float* __restrict__ att, float* __restrict__ logit,
                            float* __restrict__ amax, int ne) {
  int e = blockIdx.x * blockDim.x + threadIdx.x;
  if (e >= ne) return;
  int s = src[e], d = dst[e];
  float a = ea[e];
  const float4* ps = (const float4*)(xl + (size_t)s * 128);
  const float4* pd = (const float4*)(xr + (size_t)d * 128);
  const float4* wv = (const float4*)We;
  const float4* tv = (const float4*)att;
  __builtin_prefetch(ps, 0, 0);
  __builtin_prefetch(pd, 0, 0);
#pragma unroll
  for (int h = 0; h < 4; ++h) {
    float acc = 0.0f;
#pragma unroll
    for (int q = 0; q < 8; ++q) {
      int i = h * 8 + q;
      float4 l = ps[i], r = pd[i], w = wv[i], t = tv[i];
      float m0 = l.x + r.x + a * w.x; m0 = (m0 > 0.f) ? m0 : NEG_SLOPE * m0;
      float m1 = l.y + r.y + a * w.y; m1 = (m1 > 0.f) ? m1 : NEG_SLOPE * m1;
      float m2 = l.z + r.z + a * w.z; m2 = (m2 > 0.f) ? m2 : NEG_SLOPE * m2;
      float m3 = l.w + r.w + a * w.w; m3 = (m3 > 0.f) ? m3 : NEG_SLOPE * m3;
      acc += m0 * t.x + m1 * t.y + m2 * t.z + m3 * t.w;
    }
    logit[(size_t)e * 4 + h] = acc;
    atomic_max_float(&amax[(size_t)d * 4 + h], acc);
  }
}

__global__ void edge_exp1(const int* __restrict__ dst, const float* __restrict__ amax,
                          float* __restrict__ exv, float* __restrict__ denom, int ne) {
  int e = blockIdx.x * blockDim.x + threadIdx.x;
  if (e >= ne) return;
  int d = dst[e];
#pragma unroll
  for (int h = 0; h < 4; ++h) {
    float ex = expf(exv[(size_t)e * 4 + h] - amax[(size_t)d * 4 + h]);
    exv[(size_t)e * 4 + h] = ex;
    atomicAdd(&denom[(size_t)d * 4 + h], ex);
  }
}

// wave-per-edge: 32 lanes share one (src,dst,alpha) fetch, 4 coalesced atomics each
__global__ void edge_agg1(const int* __restrict__ src, const int* __restrict__ dst,
                          const float* __restrict__ exv, const float* __restrict__ denom,
                          const float* __restrict__ xl, float* __restrict__ agg, int ne) {
  long idx = (long)blockIdx.x * blockDim.x + threadIdx.x;
  int e = (int)(idx >> 5);
  int lane = (int)(idx & 31);
  if (e >= ne) return;
  int s = src[e], d = dst[e];
  const float* ps = xl + (size_t)s * 128;
  float* pg = agg + (size_t)d * 128;
  float al[4];
#pragma unroll
  for (int h = 0; h < 4; ++h)
    al[h] = exv[(size_t)e * 4 + h] / (denom[(size_t)d * 4 + h] + 1e-16f);
#pragma unroll
  for (int q = 0; q < 4; ++q) {       // i = lane + 32q  ->  head h == q
    int i = lane + 32 * q;
    atomicAdd(&pg[i], al[q] * ps[i]);
  }
}

// mean over 4 heads + bias -> LN(g1,beta1) -> ELU   (one 32-lane wave / node)
__global__ void finish1(const float* __restrict__ agg, const float* __restrict__ bias,
                        const float* __restrict__ g, const float* __restrict__ beta,
                        float* __restrict__ hout, int n) {
  long idx = (long)blockIdx.x * blockDim.x + threadIdx.x;
  int node = (int)(idx >> 5);
  int c = (int)(idx & 31);
  if (node >= n) return;
  const float* p = agg + (size_t)node * 128;
  float v = 0.25f * (p[c] + p[32 + c] + p[64 + c] + p[96 + c]) + bias[c];
  float s = v;
#pragma unroll
  for (int o = 16; o > 0; o >>= 1) s += __shfl_xor(s, o, 32);
  float mu = s * (1.0f / 32.0f);
  float dv = v - mu;
  float q = dv * dv;
#pragma unroll
  for (int o = 16; o > 0; o >>= 1) q += __shfl_xor(q, o, 32);
  float y = dv * rsqrtf(q * (1.0f / 32.0f) + LN_EPS) * g[c] + beta[c];
  y = (y > 0.0f) ? y : (expf(y) - 1.0f);            // ELU
  hout[(size_t)node * 32 + c] = y;
}

// ------------------------- layer 2 edge phases (H=1, C=16) -----------------
__global__ void edge_logit2(const float* __restrict__ xl, const float* __restrict__ xr,
                            const float* __restrict__ ea, const int* __restrict__ src,
                            const int* __restrict__ dst, const float* __restrict__ We,
                            const float* __restrict__ att, float* __restrict__ logit,
                            float* __restrict__ amax, int ne) {
  int e = blockIdx.x * blockDim.x + threadIdx.x;
  if (e >= ne) return;
  int s = src[e], d = dst[e];
  float a = ea[e];
  const float4* ps = (const float4*)(xl + (size_t)s * 16);
  const float4* pd = (const float4*)(xr + (size_t)d * 16);
  const float4* wv = (const float4*)We;
  const float4* tv = (const float4*)att;
  __builtin_prefetch(ps, 0, 0);
  __builtin_prefetch(pd, 0, 0);
  float acc = 0.0f;
#pragma unroll
  for (int q = 0; q < 4; ++q) {
    float4 l = ps[q], r = pd[q], w = wv[q], t = tv[q];
    float m0 = l.x + r.x + a * w.x; m0 = (m0 > 0.f) ? m0 : NEG_SLOPE * m0;
    float m1 = l.y + r.y + a * w.y; m1 = (m1 > 0.f) ? m1 : NEG_SLOPE * m1;
    float m2 = l.z + r.z + a * w.z; m2 = (m2 > 0.f) ? m2 : NEG_SLOPE * m2;
    float m3 = l.w + r.w + a * w.w; m3 = (m3 > 0.f) ? m3 : NEG_SLOPE * m3;
    acc += m0 * t.x + m1 * t.y + m2 * t.z + m3 * t.w;
  }
  logit[e] = acc;
  atomic_max_float(&amax[d], acc);
}

__global__ void edge_exp2(const int* __restrict__ dst, const float* __restrict__ amax,
                          float* __restrict__ exv, float* __restrict__ denom, int ne) {
  int e = blockIdx.x * blockDim.x + threadIdx.x;
  if (e >= ne) return;
  int d = dst[e];
  float ex = expf(exv[e] - amax[d]);
  exv[e] = ex;
  atomicAdd(&denom[d], ex);
}

__global__ void edge_agg2(const int* __restrict__ src, const int* __restrict__ dst,
                          const float* __restrict__ exv, const float* __restrict__ denom,
                          const float* __restrict__ xl, float* __restrict__ agg, int ne) {
  long idx = (long)blockIdx.x * blockDim.x + threadIdx.x;
  int e = (int)(idx >> 4);
  int c = (int)(idx & 15);
  if (e >= ne) return;
  int s = src[e], d = dst[e];
  float alpha = exv[e] / (denom[d] + 1e-16f);
  atomicAdd(&agg[(size_t)d * 16 + c], alpha * xl[(size_t)s * 16 + c]);
}

// agg + bias2 -> LN(g2,beta2) -> d_out   (16-lane subgroup per node)
__global__ void finish2(const float* __restrict__ agg, const float* __restrict__ bias,
                        const float* __restrict__ g, const float* __restrict__ beta,
                        float* __restrict__ out, int n) {
  long idx = (long)blockIdx.x * blockDim.x + threadIdx.x;
  int node = (int)(idx >> 4);
  int c = (int)(idx & 15);
  if (node >= n) return;
  float v = agg[(size_t)node * 16 + c] + bias[c];
  float s = v;
#pragma unroll
  for (int o = 8; o > 0; o >>= 1) s += __shfl_xor(s, o, 16);
  float mu = s * (1.0f / 16.0f);
  float dv = v - mu;
  float q = dv * dv;
#pragma unroll
  for (int o = 8; o > 0; o >>= 1) q += __shfl_xor(q, o, 16);
  out[(size_t)node * 16 + c] = dv * rsqrtf(q * (1.0f / 16.0f) + LN_EPS) * g[c] + beta[c];
}

// ---------------------------------------------------------------------------
extern "C" void kernel_launch(void* const* d_in, const int* in_sizes, int n_in,
                              void* d_out, int out_size, void* d_ws, size_t ws_size,
                              hipStream_t stream) {
  (void)n_in; (void)out_size; (void)ws_size;
  const float* x    = (const float*)d_in[0];
  const int*   ei   = (const int*)d_in[1];
  const float* ea   = (const float*)d_in[2];
  const float* W1l  = (const float*)d_in[3];
  const float* b1l  = (const float*)d_in[4];
  const float* W1r  = (const float*)d_in[5];
  const float* b1r  = (const float*)d_in[6];
  const float* We1  = (const float*)d_in[7];
  const float* att1 = (const float*)d_in[8];
  const float* bia1 = (const float*)d_in[9];
  const float* g1   = (const float*)d_in[10];
  const float* be1  = (const float*)d_in[11];
  const float* W2l  = (const float*)d_in[12];
  const float* b2l  = (const float*)d_in[13];
  const float* W2r  = (const float*)d_in[14];
  const float* b2r  = (const float*)d_in[15];
  const float* We2  = (const float*)d_in[16];
  const float* att2 = (const float*)d_in[17];
  const float* bia2 = (const float*)d_in[18];
  const float* g2   = (const float*)d_in[19];
  const float* be2  = (const float*)d_in[20];

  const int n  = in_sizes[0] / 2;   // N nodes (x is N x 2)
  const int ne = in_sizes[2];       // E edges (edge_attr is E x 1)
  const int* srcv = ei;
  const int* dstv = ei + ne;
  const size_t np = (size_t)n + ROWPAD;   // padded row count for WMMA outputs

  // workspace layout (regions reused across layers; xl/xr regions row-padded)
  float* ws    = (float*)d_ws;
  float* xl1   = ws;                         // np*128
  float* xr1   = xl1 + np * 128;             // np*128
  float* agg1  = xr1 + np * 128;             // n*128
  float* ex1   = agg1 + (size_t)n * 128;     // ne*4
  float* amax1 = ex1 + (size_t)ne * 4;       // n*4
  float* den1  = amax1 + (size_t)n * 4;      // n*4
  float* hbuf  = xl1;                        // n*32   (xl1 dead after edge_agg1)
  float* xl2   = xr1;                        // np*16  (xr1 dead after edge_logit1)
  float* xr2   = xr1 + np * 16;              // np*16
  float* agg2  = agg1;                       // n*16
  float* ex2   = ex1;                        // ne
  float* amax2 = amax1;                      // n
  float* den2  = den1;                       // n

  auto cdiv = [](long a, long b) { return (int)((a + b - 1) / b); };
  const float ninf = -__builtin_inff();

  // ---- layer 1 ----
  {
    int waves = cdiv(n, 16);
    int thr = waves * 32;
    xform_wmma<2, 128><<<cdiv(thr, 256), 256, 0, stream>>>(x, W1l, b1l, xl1, n);
    xform_wmma<2, 128><<<cdiv(thr, 256), 256, 0, stream>>>(x, W1r, b1r, xr1, n);
  }
  fill_f32<<<cdiv((long)n * 4, 256), 256, 0, stream>>>(amax1, ninf, (long)n * 4);
  fill_f32<<<cdiv((long)n * 4, 256), 256, 0, stream>>>(den1, 0.0f, (long)n * 4);
  fill_f32<<<cdiv((long)n * 128, 256), 256, 0, stream>>>(agg1, 0.0f, (long)n * 128);

  edge_logit1<<<cdiv(ne, 256), 256, 0, stream>>>(xl1, xr1, ea, srcv, dstv, We1, att1, ex1, amax1, ne);
  edge_exp1<<<cdiv(ne, 256), 256, 0, stream>>>(dstv, amax1, ex1, den1, ne);
  edge_agg1<<<cdiv((long)ne * 32, 256), 256, 0, stream>>>(srcv, dstv, ex1, den1, xl1, agg1, ne);
  finish1<<<cdiv((long)n * 32, 256), 256, 0, stream>>>(agg1, bia1, g1, be1, hbuf, n);

  // ---- layer 2 ----
  {
    int waves = cdiv(n, 16);
    int thr = waves * 32;
    xform_wmma<32, 16><<<cdiv(thr, 256), 256, 0, stream>>>(hbuf, W2l, b2l, xl2, n);
    xform_wmma<32, 16><<<cdiv(thr, 256), 256, 0, stream>>>(hbuf, W2r, b2r, xr2, n);
  }
  fill_f32<<<cdiv((long)n, 256), 256, 0, stream>>>(amax2, ninf, (long)n);
  fill_f32<<<cdiv((long)n, 256), 256, 0, stream>>>(den2, 0.0f, (long)n);
  fill_f32<<<cdiv((long)n * 16, 256), 256, 0, stream>>>(agg2, 0.0f, (long)n * 16);

  edge_logit2<<<cdiv(ne, 256), 256, 0, stream>>>(xl2, xr2, ea, srcv, dstv, We2, att2, ex2, amax2, ne);
  edge_exp2<<<cdiv(ne, 256), 256, 0, stream>>>(dstv, amax2, ex2, den2, ne);
  edge_agg2<<<cdiv((long)ne * 16, 256), 256, 0, stream>>>(srcv, dstv, ex2, den2, xl2, agg2, ne);
  finish2<<<cdiv((long)n * 16, 256), 256, 0, stream>>>(agg2, bia2, g2, be2, (float*)d_out, n);
}
